// RefactAttention_9045201125593
// MI455X (gfx1250) — compile-verified, matrix operand order checked by reading, not run on hardware
//
#include <hip/hip_runtime.h>

// ---------------------------------------------------------------------------
// MQA attention layer for MI455X (gfx1250, wave32, WMMA + TDM).
// fp32 math emulated with bf16 hi/lo split: each fp32 MAC tile = 3x
// v_wmma_f32_16x16x32_bf16 (hi*hi + hi*lo + lo*hi), fp32 accumulate.
// GEMMs stage block tiles into LDS with tensor_load_to_lds (TDM, TENSORcnt).
// ---------------------------------------------------------------------------

typedef float        v8f   __attribute__((ext_vector_type(8)));
typedef __bf16       v16bf __attribute__((ext_vector_type(16)));
typedef __bf16       v8bf  __attribute__((ext_vector_type(8)));
typedef unsigned int v4u   __attribute__((ext_vector_type(4)));
typedef int          v8i   __attribute__((ext_vector_type(8)));
typedef int          v4i   __attribute__((ext_vector_type(4)));

#define NUM_HEADS 32
#define HEAD_DIM  128
#define HIDDEN    4096
#define NQKV      (HIDDEN + 2 * HEAD_DIM)   // 4352
#define BATCH     2
#define SEQ       2048
#define MROWS     (BATCH * SEQ)             // 4096
#define NEG_INF   (-1.0e30f)

// LDS tile geometry for the TDM-staged GEMM: rows of 32 bf16 (64B) padded by
// TDM (pad_interval=16 dwords, pad_amount=4 dwords) to an 80B stride.
// 80B keeps every row 16B-aligned for ds_load_b128 and rotates banks by 20
// dwords/row -> conflict-free fragment reads.
#define ROW_ELEMS 40                        // 80 bytes / 2
#define A_TILE_ELEMS (64 * ROW_ELEMS)       // 2560
#define B_TILE_ELEMS (128 * ROW_ELEMS)      // 5120
#define BUF_ELEMS (2 * A_TILE_ELEMS + 2 * B_TILE_ELEMS)  // 15360 (30720 B)

// ---- fragment loaders ------------------------------------------------------
// 16-bit A (16x32) and B (32x16, from N-major "Bt") share the same per-lane
// pattern: lanes 0-15 hold K {0..7, 16..23}, lanes 16-31 hold K {8..15, 24..31}
// for row (A: M, B: N) = lane&15.  Two contiguous 16B loads per lane.
static __device__ __forceinline__ v16bf load_frag(const __bf16* __restrict__ base,
                                                  int ld, int row, int kbase, int lane) {
    const int hl = lane >> 4;
    const int r  = lane & 15;
    const __bf16* p = base + (size_t)(row + r) * (size_t)ld + (size_t)(kbase + 8 * hl);
    v8bf a = *(const v8bf*)(p);
    v8bf b = *(const v8bf*)(p + 16);
    return __builtin_shufflevector(a, b, 0, 1, 2, 3, 4, 5, 6, 7,
                                         8, 9, 10, 11, 12, 13, 14, 15);
}

static __device__ __forceinline__ v16bf load_frag_lds(const __bf16* base, int row, int lane) {
    const int hl = lane >> 4;
    const int r  = lane & 15;
    const __bf16* p = base + (row + r) * ROW_ELEMS + 8 * hl;
    v8bf a = *(const v8bf*)(p);
    v8bf b = *(const v8bf*)(p + 16);
    return __builtin_shufflevector(a, b, 0, 1, 2, 3, 4, 5, 6, 7,
                                         8, 9, 10, 11, 12, 13, 14, 15);
}

// bf16x3: acc += (ah+al)*(bh+bl), dropping the lo*lo term (~2^-16 rel err)
static __device__ __forceinline__ v8f wmma3(v16bf ah, v16bf al, v16bf bh, v16bf bl, v8f acc) {
    acc = __builtin_amdgcn_wmma_f32_16x16x32_bf16(false, al, false, bh, (short)0, acc, false, false);
    acc = __builtin_amdgcn_wmma_f32_16x16x32_bf16(false, ah, false, bl, (short)0, acc, false, false);
    acc = __builtin_amdgcn_wmma_f32_16x16x32_bf16(false, ah, false, bh, (short)0, acc, false, false);
    return acc;
}

static __device__ __forceinline__ void split_bf16(float x, __bf16& hi, __bf16& lo) {
    hi = (__bf16)x;
    lo = (__bf16)(x - (float)hi);
}

// ---- TDM: DMA a (rows x 32) bf16 tile, row stride ld_elems, into LDS -------
// D# per CDNA5 ISA ch.8: group0 = {count|flags, lds_addr, gaddr_lo, gaddr_hi|type},
// group1 = dims/strides/tile/pad.  2-D tensor -> groups 2/3 zero (disabled).
// Toolchain uses the 6-arg builtin: (g0, g1, g2, g3, extra_group, cpol).
static __device__ __forceinline__ void tdm_load_tile(unsigned lds_addr,
                                                     const __bf16* gaddr,
                                                     int rows, int ld_elems) {
    unsigned long long ga = (unsigned long long)(size_t)gaddr;
    v4u g0;
    g0[0] = 1u;                                          // count=1 (valid), user mode
    g0[1] = lds_addr;                                    // LDS byte address
    g0[2] = (unsigned)(ga & 0xffffffffu);                // global_addr[31:0]
    g0[3] = (unsigned)((ga >> 32) & 0x1ffffffu) | (2u << 30);  // addr[56:32] | type=2
    v8i g1;
    g1[0] = (1 << 16)                                    // data_size = 1 -> 2 bytes
          | (1 << 20)                                    // pad_enable
          | (3 << 22)                                    // pad_interval code 3 = 16 dwords (one row)
          | (3 << 25);                                   // pad_amount  code 3 = 4 dwords (16 B)
    g1[1] = (32 << 16);                                  // tensor_dim0[15:0] = 32 elems
    g1[2] = (rows & 0xffff) << 16;                       // tensor_dim1[15:0] = rows
    g1[3] = (32 << 16);                                  // tile_dim0 = 32 elems
    g1[4] = (rows & 0xffff);                             // tile_dim1 = rows, tile_dim2 = 0
    g1[5] = ld_elems;                                    // tensor_dim0_stride[31:0]
    g1[6] = 0;
    g1[7] = 0;
    v4i z4 = {0, 0, 0, 0};
    v8i z8 = {0, 0, 0, 0, 0, 0, 0, 0};
    __builtin_amdgcn_tensor_load_to_lds(g0, g1, z4, z4, z8, 0);
}

// ---- prep kernels ----------------------------------------------------------
__global__ void split_plain_kernel(const float* __restrict__ in,
                                   __bf16* __restrict__ hi, __bf16* __restrict__ lo,
                                   size_t n) {
    size_t i = (size_t)blockIdx.x * blockDim.x + threadIdx.x;
    if (i >= n) return;
    __bf16 h, l;
    split_bf16(in[i], h, l);
    hi[i] = h; lo[i] = l;
}

// in: R x C row-major  ->  out: C x R row-major (bf16 hi/lo planes)
__global__ void split_transpose_kernel(const float* __restrict__ in,
                                       __bf16* __restrict__ hi, __bf16* __restrict__ lo,
                                       int R, int C) {
    int c = blockIdx.x * blockDim.x + threadIdx.x;
    int r = blockIdx.y;
    if (c >= C) return;
    __bf16 h, l;
    split_bf16(in[(size_t)r * C + c], h, l);
    size_t o = (size_t)c * R + r;
    hi[o] = h; lo[o] = l;
}

// qkv (MROWS x 4352) -> Q planes (MROWS x 4096), K planes (MROWS x 128),
// Vt planes (B x 128 x SEQ, transposed so PV B-fragments are contiguous)
__global__ void qkv_split_kernel(const float* __restrict__ qkv,
                                 __bf16* __restrict__ Qhi, __bf16* __restrict__ Qlo,
                                 __bf16* __restrict__ Khi, __bf16* __restrict__ Klo,
                                 __bf16* __restrict__ Vthi, __bf16* __restrict__ Vtlo) {
    int c   = blockIdx.x * blockDim.x + threadIdx.x;
    int row = blockIdx.y;                     // b*SEQ + s
    if (c >= NQKV) return;
    __bf16 h, l;
    split_bf16(qkv[(size_t)row * NQKV + c], h, l);
    if (c < HIDDEN) {
        size_t o = (size_t)row * HIDDEN + c;
        Qhi[o] = h; Qlo[o] = l;
    } else if (c < HIDDEN + HEAD_DIM) {
        size_t o = (size_t)row * HEAD_DIM + (c - HIDDEN);
        Khi[o] = h; Klo[o] = l;
    } else {
        int d = c - (HIDDEN + HEAD_DIM);
        int b = row >> 11, s = row & (SEQ - 1);
        size_t o = (size_t)b * HEAD_DIM * SEQ + (size_t)d * SEQ + s;
        Vthi[o] = h; Vtlo[o] = l;
    }
}

// ---- GEMM: C(MxN,f32) = A(MxK) * Bt^T, bf16x3, TDM-staged LDS tiles --------
// 4 waves/block as 2(M) x 2(N); wave tile 32x64; block tile 64x128.
// Double-buffered: wave 0 DMAs the next k-step's A/B hi/lo tiles into LDS with
// tensor_load_to_lds while all waves compute on the current buffer.
__global__ void __launch_bounds__(128) gemm_bf16x3_kernel(
    const __bf16* __restrict__ Ahi, const __bf16* __restrict__ Alo,
    const __bf16* __restrict__ Bthi, const __bf16* __restrict__ Btlo,
    float* __restrict__ C, int M, int N, int K) {
    __shared__ __bf16 smem[2 * BUF_ELEMS];   // 61440 B

    const int lane = threadIdx.x & 31;
    const int w    = threadIdx.x >> 5;
    const int wm   = w >> 1, wn = w & 1;
    const int mblk = blockIdx.y * 64;
    const int nblk = blockIdx.x * 128;
    const int mbase = wm * 32;               // within block tile
    const int nbase = wn * 64;
    const unsigned lds0 = (unsigned)(size_t)(&smem[0]);

    // issue one buffer's 4 tiles (A hi/lo 64x32, B hi/lo 128x32)
    auto issue = [&](int buf, int k0) {
        unsigned base = lds0 + (unsigned)(buf * BUF_ELEMS * 2);   // bytes
        tdm_load_tile(base,                          Ahi  + (size_t)mblk * K + k0,  64, K);
        tdm_load_tile(base + A_TILE_ELEMS * 2,       Alo  + (size_t)mblk * K + k0,  64, K);
        tdm_load_tile(base + A_TILE_ELEMS * 4,       Bthi + (size_t)nblk * K + k0, 128, K);
        tdm_load_tile(base + A_TILE_ELEMS * 4 + B_TILE_ELEMS * 2,
                                                     Btlo + (size_t)nblk * K + k0, 128, K);
    };

    if (w == 0) issue(0, 0);

    v8f acc[2][4] = {};
    int buf = 0;
    const int nk = K / 32;
    for (int kidx = 0; kidx < nk; kidx++) {
        if (w == 0) {
            if (kidx + 1 < nk) {
                issue(buf ^ 1, (kidx + 1) * 32);
                __builtin_amdgcn_s_wait_tensorcnt((short)4);  // current buffer done
            } else {
                __builtin_amdgcn_s_wait_tensorcnt((short)0);
            }
        }
        __syncthreads();                      // LDS tiles visible to all waves

        const __bf16* sAh = smem + buf * BUF_ELEMS;
        const __bf16* sAl = sAh + A_TILE_ELEMS;
        const __bf16* sBh = sAl + A_TILE_ELEMS;
        const __bf16* sBl = sBh + B_TILE_ELEMS;

        v16bf ah[2], al[2];
#pragma unroll
        for (int i = 0; i < 2; i++) {
            ah[i] = load_frag_lds(sAh, mbase + 16 * i, lane);
            al[i] = load_frag_lds(sAl, mbase + 16 * i, lane);
        }
#pragma unroll
        for (int j = 0; j < 4; j++) {
            v16bf bh = load_frag_lds(sBh, nbase + 16 * j, lane);
            v16bf bl = load_frag_lds(sBl, nbase + 16 * j, lane);
#pragma unroll
            for (int i = 0; i < 2; i++)
                acc[i][j] = wmma3(ah[i], al[i], bh, bl, acc[i][j]);
        }
        __syncthreads();                      // all waves done before buffer reuse
        buf ^= 1;
    }

    const int col = lane & 15, hl = lane >> 4;
#pragma unroll
    for (int i = 0; i < 2; i++)
#pragma unroll
        for (int j = 0; j < 4; j++)
#pragma unroll
            for (int r = 0; r < 8; r++) {
                int m = mblk + mbase + 16 * i + r + 8 * hl;
                int n = nblk + nbase + 16 * j + col;
                C[(size_t)m * N + n] = acc[i][j][r];
            }
    (void)M;
}

// ---- Flash attention (causal + ALiBi, 1 KV head) ---------------------------
// Block: 4 waves, each owns 16 q rows (block = 64 rows). K/V tiles of 64.
// Causal tile count is exactly blockIdx.x+1 for every wave -> uniform loop.
__global__ void __launch_bounds__(128) flash_attn_kernel(
    const __bf16* __restrict__ Qhi, const __bf16* __restrict__ Qlo,
    const __bf16* __restrict__ Khi, const __bf16* __restrict__ Klo,
    const __bf16* __restrict__ Vthi, const __bf16* __restrict__ Vtlo,
    float* __restrict__ attn) {
    __shared__ float Plds[4][16 * 65];        // padded: stride 65 -> no bank conflicts

    const int lane  = threadIdx.x & 31;
    const int w     = threadIdx.x >> 5;
    const int b     = blockIdx.z;
    const int h     = blockIdx.y;
    const int qbase = blockIdx.x * 64 + w * 16;      // first q row (seq pos) of this wave
    const int col   = lane & 15, hl = lane >> 4;
    const int row0  = b * SEQ + qbase;
    const float scale = 0.08838834764831843f;        // 128^-0.5
    const float slope = exp2f(-0.25f * (float)(h + 1));

    // preload Q fragments (16 x 128 as 4 k-steps, hi+lo)
    v16bf qh[4], ql[4];
#pragma unroll
    for (int ks = 0; ks < 4; ks++) {
        qh[ks] = load_frag(Qhi, HIDDEN, row0, h * HEAD_DIM + ks * 32, lane);
        ql[ks] = load_frag(Qlo, HIDDEN, row0, h * HEAD_DIM + ks * 32, lane);
    }

    v8f O[8] = {};
    float mi[8], li[8];
#pragma unroll
    for (int r = 0; r < 8; r++) { mi[r] = -3.0e38f; li[r] = 0.0f; }

    const __bf16* Vthb = Vthi + (size_t)b * HEAD_DIM * SEQ;
    const __bf16* Vtlb = Vtlo + (size_t)b * HEAD_DIM * SEQ;

    const int ktmax = blockIdx.x;                    // uniform across the block
    for (int kt = 0; kt <= ktmax; kt++) {
        const int kb = kt * 64;

        // S = Q * K^T  (16 x 64 as 4 tiles)
        v8f s[4];
#pragma unroll
        for (int t = 0; t < 4; t++) {
            v8f sa = {};
#pragma unroll
            for (int ks = 0; ks < 4; ks++) {
                v16bf bh = load_frag(Khi, HEAD_DIM, b * SEQ + kb + 16 * t, ks * 32, lane);
                v16bf bl = load_frag(Klo, HEAD_DIM, b * SEQ + kb + 16 * t, ks * 32, lane);
                sa = wmma3(qh[ks], ql[ks], bh, bl, sa);
            }
            s[t] = sa;
        }

        // scale + ALiBi + causal mask, online softmax
#pragma unroll
        for (int r = 0; r < 8; r++) {
            const int qg = qbase + r + 8 * hl;
            float rm = -3.0e38f;
#pragma unroll
            for (int t = 0; t < 4; t++) {
                int ng = kb + 16 * t + col;
                float x = s[t][r];
                x = (ng <= qg) ? x * scale + slope * (float)(ng - qg) : NEG_INF;
                s[t][r] = x;
                rm = fmaxf(rm, x);
            }
            rm = fmaxf(rm, __shfl_xor(rm, 1));
            rm = fmaxf(rm, __shfl_xor(rm, 2));
            rm = fmaxf(rm, __shfl_xor(rm, 4));
            rm = fmaxf(rm, __shfl_xor(rm, 8));
            const float mnew = fmaxf(mi[r], rm);
            const float corr = __expf(mi[r] - mnew);
            mi[r] = mnew;
            li[r] *= corr;
#pragma unroll
            for (int nt = 0; nt < 8; nt++) O[nt][r] *= corr;
            float rs = 0.0f;
#pragma unroll
            for (int t = 0; t < 4; t++) {
                float p = __expf(s[t][r] - mnew);
                rs += p;
                Plds[w][(r + 8 * hl) * 65 + 16 * t + col] = p;
            }
            rs += __shfl_xor(rs, 1);
            rs += __shfl_xor(rs, 2);
            rs += __shfl_xor(rs, 4);
            rs += __shfl_xor(rs, 8);
            li[r] += rs;
        }

        __syncthreads();   // uniform; orders LDS P writes before fragment reads

        // rebuild P (16x64) as A-fragments (hi/lo) from LDS
        v16bf ph[2], pl[2];
#pragma unroll
        for (int ks = 0; ks < 2; ks++) {
            const int mrow = lane & 15;
            const int k0 = 32 * ks + 8 * hl;
#pragma unroll
            for (int e = 0; e < 16; e++) {
                int kk = (e < 8) ? (k0 + e) : (k0 + 16 + (e - 8));
                float pv = Plds[w][mrow * 65 + kk];
                __bf16 hh = (__bf16)pv;
                ph[ks][e] = hh;
                pl[ks][e] = (__bf16)(pv - (float)hh);
            }
        }

        // O += P * V   (16x64 @ 64x128)
#pragma unroll
        for (int nt = 0; nt < 8; nt++) {
#pragma unroll
            for (int ks = 0; ks < 2; ks++) {
                v16bf vh = load_frag(Vthb, SEQ, 16 * nt, kb + 32 * ks, lane);
                v16bf vl = load_frag(Vtlb, SEQ, 16 * nt, kb + 32 * ks, lane);
                O[nt] = wmma3(ph[ks], pl[ks], vh, vl, O[nt]);
            }
        }
        __syncthreads();   // protect Plds before next iteration rewrites it
    }

    // normalize + write attn[b, s, h*128 + d]
#pragma unroll
    for (int r = 0; r < 8; r++) {
        const float inv = 1.0f / li[r];
        const size_t rowg = (size_t)(row0 + r + 8 * hl);
#pragma unroll
        for (int nt = 0; nt < 8; nt++)
            attn[rowg * HIDDEN + h * HEAD_DIM + 16 * nt + col] = O[nt][r] * inv;
    }
}

// ---------------------------------------------------------------------------
extern "C" void kernel_launch(void* const* d_in, const int* in_sizes, int n_in,
                              void* d_out, int out_size, void* d_ws, size_t ws_size,
                              hipStream_t stream) {
    const float* hidden = (const float*)d_in[0];   // (2, 2048, 4096)
    const float* Wqkv   = (const float*)d_in[1];   // (4096, 4352)
    const float* Wo     = (const float*)d_in[2];   // (4096, 4096)
    float* out = (float*)d_out;                    // (2, 2048, 4096) f32
    char* ws = (char*)d_ws;

    // workspace layout (bytes); total ~396 MiB
    const size_t SZ_XPL  = (size_t)MROWS * HIDDEN * 2;          // 32 MiB / plane
    const size_t SZ_WQPL = (size_t)NQKV * HIDDEN * 2;           // 34 MiB / plane
    const size_t SZ_WOPL = (size_t)HIDDEN * HIDDEN * 2;         // 32 MiB / plane
    const size_t SZ_QKV  = (size_t)MROWS * NQKV * 4;            // 68 MiB
    const size_t SZ_KVPL = (size_t)BATCH * SEQ * HEAD_DIM * 2;  // 1 MiB / plane

    size_t off = 0;
    __bf16* Xhi   = (__bf16*)(ws + off); off += SZ_XPL;
    __bf16* Xlo   = (__bf16*)(ws + off); off += SZ_XPL;
    __bf16* WqThi = (__bf16*)(ws + off); off += SZ_WQPL;
    __bf16* WqTlo = (__bf16*)(ws + off); off += SZ_WQPL;
    __bf16* WoThi = (__bf16*)(ws + off); off += SZ_WOPL;
    __bf16* WoTlo = (__bf16*)(ws + off); off += SZ_WOPL;
    float*  qkv   = (float*) (ws + off); off += SZ_QKV;
    __bf16* Qhi   = (__bf16*)(ws + off); off += SZ_XPL;
    __bf16* Qlo   = (__bf16*)(ws + off); off += SZ_XPL;
    __bf16* Khi   = (__bf16*)(ws + off); off += SZ_KVPL;
    __bf16* Klo   = (__bf16*)(ws + off); off += SZ_KVPL;
    __bf16* Vthi  = (__bf16*)(ws + off); off += SZ_KVPL;
    __bf16* Vtlo  = (__bf16*)(ws + off); off += SZ_KVPL;
    float*  attn  = (float*) (ws + off); off += (size_t)MROWS * HIDDEN * 4;
    // X planes are dead after GEMM1 -> reuse for attn planes
    __bf16* Ahi = Xhi;
    __bf16* Alo = Xlo;
    (void)ws_size; (void)in_sizes; (void)n_in; (void)out_size;

    // 1) split hidden -> bf16 hi/lo
    {
        size_t n = (size_t)MROWS * HIDDEN;
        split_plain_kernel<<<dim3((unsigned)((n + 255) / 256)), 256, 0, stream>>>(hidden, Xhi, Xlo, n);
    }
    // 2) split+transpose weights
    split_transpose_kernel<<<dim3((NQKV + 255) / 256, HIDDEN), 256, 0, stream>>>(Wqkv, WqThi, WqTlo, HIDDEN, NQKV);
    split_transpose_kernel<<<dim3((HIDDEN + 255) / 256, HIDDEN), 256, 0, stream>>>(Wo, WoThi, WoTlo, HIDDEN, HIDDEN);
    // 3) qkv = X @ Wqkv   (4096 x 4352, K=4096), block tile 64x128
    gemm_bf16x3_kernel<<<dim3(NQKV / 128, MROWS / 64), 128, 0, stream>>>(
        Xhi, Xlo, WqThi, WqTlo, qkv, MROWS, NQKV, HIDDEN);
    // 4) split qkv -> Q/K/Vt planes
    qkv_split_kernel<<<dim3((NQKV + 255) / 256, MROWS), 256, 0, stream>>>(
        qkv, Qhi, Qlo, Khi, Klo, Vthi, Vtlo);
    // 5) flash attention
    flash_attn_kernel<<<dim3(SEQ / 64, NUM_HEADS, BATCH), 128, 0, stream>>>(
        Qhi, Qlo, Khi, Klo, Vthi, Vtlo, attn);
    // 6) split attn -> bf16 hi/lo (reusing X plane storage)
    {
        size_t n = (size_t)MROWS * HIDDEN;
        split_plain_kernel<<<dim3((unsigned)((n + 255) / 256)), 256, 0, stream>>>(attn, Ahi, Alo, n);
    }
    // 7) out = attn @ Wo   (4096 x 4096, K=4096)
    gemm_bf16x3_kernel<<<dim3(HIDDEN / 128, MROWS / 64), 128, 0, stream>>>(
        Ahi, Alo, WoThi, WoTlo, out, MROWS, HIDDEN, HIDDEN);
}
